// GravityAttention_79774722556516
// MI455X (gfx1250) — matrix operand order repaired and checked by reading, not compiled, not run on hardware
//
#include <hip/hip_runtime.h>
#include <hip/hip_bf16.h>

typedef __attribute__((ext_vector_type(16))) _Float16 v16h;
typedef __attribute__((ext_vector_type(8)))  _Float16 v8h;
typedef __attribute__((ext_vector_type(8)))  float    v8f;

#define B_  4
#define T_  2048
#define C_  1024
#define H_  16
#define D_  64

// ---------------------------------------------------------------------------
// WMMA helpers (CDNA5 16x16x32 f16 -> f32)
// ---------------------------------------------------------------------------
__device__ __forceinline__ v8f wmma16(v16h a, v16h b, v8f c) {
    return __builtin_amdgcn_wmma_f32_16x16x32_f16(
        /*neg_a=*/false, a, /*neg_b=*/false, b,
        /*c_mod=*/(short)0, c, /*reuse_a=*/false, /*reuse_b=*/false);
}

// A fragment: 16(M) x 32(K) f16 from row-major src (ld in halves).
__device__ __forceinline__ v16h load_a(const _Float16* base, int ld, int k0) {
    int lane = threadIdx.x & 31;
    const _Float16* p = base + (size_t)(lane & 15) * ld + k0 + ((lane >> 4) << 3);
    union { v16h v; v8h h[2]; } u;
    u.h[0] = *(const v8h*)(p);
    u.h[1] = *(const v8h*)(p + 16);
    return u.v;
}

// B fragment: 32(K) x 16(N) f16; column n of B = row n of BT (row-major).
__device__ __forceinline__ v16h load_b(const _Float16* base, int ld, int k0) {
    int lane = threadIdx.x & 31;
    const _Float16* p = base + (size_t)(lane & 15) * ld + k0 + ((lane >> 4) << 3);
    union { v16h v; v8h h[2]; } u;
    u.h[0] = *(const v8h*)(p);
    u.h[1] = *(const v8h*)(p + 16);
    return u.v;
}

// One wave computes a 32(M) x 64(N) tile: two rows of 4 C-tiles; the 4 B
// fragments per k-step are shared by both A fragments (8 WMMAs / 6 loads).
__device__ __forceinline__ void gemm_tile2(const _Float16* __restrict__ A,
                                           const _Float16* __restrict__ BT,
                                           int K, int lda, int ldb,
                                           v8f acc0[4], v8f acc1[4]) {
    for (int k = 0; k < K; k += 32) {
        v16h a0 = load_a(A, lda, k);
        v16h a1 = load_a(A + (size_t)16 * lda, lda, k);
#pragma unroll
        for (int j = 0; j < 4; ++j) {
            v16h b = load_b(BT + (size_t)j * 16 * ldb, ldb, k);
            acc0[j] = wmma16(a0, b, acc0[j]);
            acc1[j] = wmma16(a1, b, acc1[j]);
        }
    }
}

// ---------------------------------------------------------------------------
// Elementwise prep kernels
// ---------------------------------------------------------------------------
__global__ void convert_f32_to_f16(const float* __restrict__ in,
                                   _Float16* __restrict__ out, int n) {
    int i = blockIdx.x * blockDim.x + threadIdx.x;
    if (i < n) out[i] = (_Float16)in[i];
}

// out[n*K + k] = (f16) in[k*N + n]   (transpose KxN -> NxK)
__global__ void transpose_to_f16(const float* __restrict__ in,
                                 _Float16* __restrict__ out, int K, int N) {
    int i = blockIdx.x * blockDim.x + threadIdx.x;
    if (i < K * N) {
        int n = i / K, k = i - n * K;
        out[i] = (_Float16)in[(size_t)k * N + n];
    }
}

// ---------------------------------------------------------------------------
// QKV GEMM + RoPE epilogue (per 16x64 accumulator group).
// ---------------------------------------------------------------------------
__device__ __forceinline__ void qkv_epilogue(v8f acc[4], int m0, int n0,
                                             _Float16* __restrict__ qh,
                                             _Float16* __restrict__ kh,
                                             _Float16* __restrict__ vT) {
    int lane = threadIdx.x & 31, c = lane & 15, hi = lane >> 4;
    int b  = m0 >> 11;              // / 2048
    int t0 = m0 & (T_ - 1);
    int which = n0 >> 10;           // 0=q 1=k 2=v
    int h  = (n0 >> 6) & (H_ - 1);

    if (which < 2) {
        // RoPE: pair (d1, d1+32) lives in tiles (j, j+2), same lane/vgpr.
#pragma unroll
        for (int j = 0; j < 2; ++j) {
            int d1 = (j << 4) + c;                       // 0..31
            float inv = __expf(-0.28782313662425572f * (float)d1); // 10000^(-d1/32)
#pragma unroll
            for (int v = 0; v < 8; ++v) {
                int t = t0 + v + (hi << 3);
                float sn, cs;
                __sincosf((float)t * inv, &sn, &cs);
                float x1 = acc[j][v], x2 = acc[j + 2][v];
                acc[j][v]     = x1 * cs - x2 * sn;
                acc[j + 2][v] = x1 * sn + x2 * cs;
            }
        }
        float sc = (which == 0) ? 0.125f : 1.0f;         // fold 1/sqrt(d) into Q
        _Float16* dst = ((which == 0) ? qh : kh) + ((size_t)(b * H_ + h)) * T_ * D_;
#pragma unroll
        for (int j = 0; j < 4; ++j)
#pragma unroll
            for (int v = 0; v < 8; ++v)
                dst[(size_t)(t0 + v + (hi << 3)) * D_ + (j << 4) + c] =
                    (_Float16)(acc[j][v] * sc);
    } else {
        // V stored transposed [d][t]; rows v=0..7 are contiguous t -> one b128.
        _Float16* dst = vT + ((size_t)(b * H_ + h)) * D_ * T_;
#pragma unroll
        for (int j = 0; j < 4; ++j) {
            _Float16 tmp[8];
#pragma unroll
            for (int v = 0; v < 8; ++v) tmp[v] = (_Float16)acc[j][v];
            int d = (j << 4) + c;
            *(v8h*)(dst + (size_t)d * T_ + t0 + (hi << 3)) = *(const v8h*)tmp;
        }
    }
}

__global__ void __launch_bounds__(128)
qkv_rope_kernel(const _Float16* __restrict__ xh, const _Float16* __restrict__ WT,
                _Float16* __restrict__ qh, _Float16* __restrict__ kh,
                _Float16* __restrict__ vT) {
    int gw    = (blockIdx.x * blockDim.x + threadIdx.x) >> 5; // wave id
    int ntile = gw % 48;            // 3072 / 64
    int mtile = gw / 48;            // 8192 / 32
    int m0 = mtile << 5, n0 = ntile << 6;

    v8f acc0[4] = {}, acc1[4] = {};
    gemm_tile2(xh + (size_t)m0 * C_, WT + (size_t)n0 * C_, C_, C_, C_, acc0, acc1);
    qkv_epilogue(acc0, m0,      n0, qh, kh, vT);
    qkv_epilogue(acc1, m0 + 16, n0, qh, kh, vT);
}

// ---------------------------------------------------------------------------
// Flash attention: one wave per (b, h, 16-query tile), streaming 64-key chunks.
// oh layout: [B, T, H*D] f16 (== [B,T,C], ready for the output GEMM).
// ---------------------------------------------------------------------------
__global__ void __launch_bounds__(32)
attn_kernel(const _Float16* __restrict__ qh, const _Float16* __restrict__ kh,
            const _Float16* __restrict__ vT, _Float16* __restrict__ oh) {
    int bid   = blockIdx.x;
    int qtile = bid & 127;          // T/16
    int bh    = bid >> 7;           // b*H + h
    int q0    = qtile << 4;

    const _Float16* Q  = qh + (size_t)bh * T_ * D_;
    const _Float16* Kp = kh + (size_t)bh * T_ * D_;
    const _Float16* Vt = vT + (size_t)bh * D_ * T_;

    __shared__ _Float16 P[16 * 64];  // P chunk staging (C-layout -> A-layout)

    int lane = threadIdx.x & 31, c = lane & 15, hi = lane >> 4;

    v16h aq0 = load_a(Q + (size_t)q0 * D_, D_, 0);
    v16h aq1 = load_a(Q + (size_t)q0 * D_, D_, 32);

    v8f o[4] = {};
    float m[8], l[8];
#pragma unroll
    for (int v = 0; v < 8; ++v) { m[v] = -1e30f; l[v] = 0.0f; }

    int nk = q0 + 16;               // keys needed: 0 .. q0+15
    for (int kb = 0; kb < nk; kb += 64) {
        // S = Q K^T for 4 key sub-tiles (64 keys)
        v8f s[4] = {};
#pragma unroll
        for (int t = 0; t < 4; ++t) {
            v16h b0 = load_b(Kp + (size_t)(kb + t * 16) * D_, D_, 0);
            s[t] = wmma16(aq0, b0, s[t]);
            v16h b1 = load_b(Kp + (size_t)(kb + t * 16) * D_, D_, 32);
            s[t] = wmma16(aq1, b1, s[t]);
        }
        if (kb + 63 > q0) {          // causal mask (only near/after diagonal)
#pragma unroll
            for (int v = 0; v < 8; ++v) {
                int qt = q0 + v + (hi << 3);
#pragma unroll
                for (int t = 0; t < 4; ++t)
                    if (kb + t * 16 + c > qt) s[t][v] = -1e30f;
            }
        }
        // online softmax; rows live per half-wave -> width-16 reductions
#pragma unroll
        for (int v = 0; v < 8; ++v) {
            float mc = fmaxf(fmaxf(s[0][v], s[1][v]), fmaxf(s[2][v], s[3][v]));
            mc = fmaxf(mc, __shfl_xor(mc, 1, 16));
            mc = fmaxf(mc, __shfl_xor(mc, 2, 16));
            mc = fmaxf(mc, __shfl_xor(mc, 4, 16));
            mc = fmaxf(mc, __shfl_xor(mc, 8, 16));
            float mn   = fmaxf(m[v], mc);
            float corr = __expf(m[v] - mn);
            float p[4], rs = 0.0f;
#pragma unroll
            for (int t = 0; t < 4; ++t) { p[t] = __expf(s[t][v] - mn); rs += p[t]; }
            rs += __shfl_xor(rs, 1, 16);
            rs += __shfl_xor(rs, 2, 16);
            rs += __shfl_xor(rs, 4, 16);
            rs += __shfl_xor(rs, 8, 16);
            l[v] = l[v] * corr + rs;
            m[v] = mn;
#pragma unroll
            for (int j = 0; j < 4; ++j) o[j][v] *= corr;
#pragma unroll
            for (int t = 0; t < 4; ++t)
                P[(v + (hi << 3)) * 64 + t * 16 + c] = (_Float16)p[t];
        }
        // PV: A = P (16x64 -> two K=32 fragments), B = V chunk per 16-dim tile
        v16h ap0 = load_a((const _Float16*)P, 64, 0);   // DS in-order per wave
        v16h ap1 = load_a((const _Float16*)P, 64, 32);
#pragma unroll
        for (int j = 0; j < 4; ++j) {
            v16h bv0 = load_b(Vt + (size_t)(j << 4) * T_, T_, kb);
            o[j] = wmma16(ap0, bv0, o[j]);
            v16h bv1 = load_b(Vt + (size_t)(j << 4) * T_, T_, kb + 32);
            o[j] = wmma16(ap1, bv1, o[j]);
        }
    }

    int b_ = bh >> 4, h_ = bh & (H_ - 1);
#pragma unroll
    for (int v = 0; v < 8; ++v) {
        float invl = 1.0f / l[v];
        int t = q0 + v + (hi << 3);
#pragma unroll
        for (int j = 0; j < 4; ++j)
            oh[(size_t)(b_ * T_ + t) * C_ + h_ * D_ + (j << 4) + c] =
                (_Float16)(o[j][v] * invl);
    }
}

// ---------------------------------------------------------------------------
// Output projection: out = oh(8192x1024) @ Wout, via WoutT, f32 epilogue.
// ---------------------------------------------------------------------------
__global__ void __launch_bounds__(128)
out_gemm_kernel(const _Float16* __restrict__ oh, const _Float16* __restrict__ WT,
                float* __restrict__ out) {
    int gw    = (blockIdx.x * blockDim.x + threadIdx.x) >> 5;
    int ntile = gw & 15;            // 1024 / 64
    int mtile = gw >> 4;            // 8192 / 32
    int m0 = mtile << 5, n0 = ntile << 6;

    v8f acc0[4] = {}, acc1[4] = {};
    gemm_tile2(oh + (size_t)m0 * C_, WT + (size_t)n0 * C_, C_, C_, C_, acc0, acc1);

    int lane = threadIdx.x & 31, c = lane & 15, hi = lane >> 4;
#pragma unroll
    for (int j = 0; j < 4; ++j)
#pragma unroll
        for (int v = 0; v < 8; ++v) {
            out[(size_t)(m0 + v + (hi << 3)) * C_ + n0 + (j << 4) + c]      = acc0[j][v];
            out[(size_t)(m0 + 16 + v + (hi << 3)) * C_ + n0 + (j << 4) + c] = acc1[j][v];
        }
}

// ---------------------------------------------------------------------------
extern "C" void kernel_launch(void* const* d_in, const int* in_sizes, int n_in,
                              void* d_out, int out_size, void* d_ws, size_t ws_size,
                              hipStream_t stream) {
    const float* x    = (const float*)d_in[0];   // [4,2048,1024]
    const float* Wqkv = (const float*)d_in[1];   // [1024,3072]
    const float* Wout = (const float*)d_in[2];   // [1024,1024]
    float*       out  = (float*)d_out;           // [4,2048,1024]

    const size_t NX = (size_t)B_ * T_ * C_;      // 8388608
    _Float16* xh    = (_Float16*)d_ws;
    _Float16* wqkvT = xh    + NX;                // 3072*1024
    _Float16* woutT = wqkvT + (size_t)3 * C_ * C_;
    _Float16* qh    = woutT + (size_t)C_ * C_;   // [B,H,T,D]
    _Float16* kh    = qh    + NX;
    _Float16* vT    = kh    + NX;                // [B,H,D,T]
    _Float16* oh    = vT    + NX;                // [B,T,C]

    // 1) f16 conversions / weight transposes
    convert_f32_to_f16<<<(int)((NX + 255) / 256), 256, 0, stream>>>(x, xh, (int)NX);
    transpose_to_f16<<<(3 * C_ * C_ + 255) / 256, 256, 0, stream>>>(Wqkv, wqkvT, C_, 3 * C_);
    transpose_to_f16<<<(C_ * C_ + 255) / 256, 256, 0, stream>>>(Wout, woutT, C_, C_);

    // 2) QKV GEMM + RoPE (256 mtiles x 48 ntiles waves, 4 waves/block)
    qkv_rope_kernel<<<(256 * 48) / 4, 128, 0, stream>>>(xh, wqkvT, qh, kh, vT);

    // 3) flash attention: B*H*(T/16) = 8192 single-wave blocks
    attn_kernel<<<B_ * H_ * (T_ / 16), 32, 0, stream>>>(qh, kh, vT, oh);

    // 4) output projection (256 x 16 waves)
    out_gemm_kernel<<<(256 * 16) / 4, 128, 0, stream>>>(oh, woutT, out);
}
